// AutoencoderKL_90494960927432
// MI455X (gfx1250) — compile-verified, matrix operand order checked by reading, not run on hardware
//
#include <hip/hip_runtime.h>
#include <hip/hip_bf16.h>
#include <math.h>

typedef __attribute__((ext_vector_type(16))) _Float16 v16h;
typedef __attribute__((ext_vector_type(8)))  float    v8f;
typedef __attribute__((ext_vector_type(4)))  int      v4i;

#define AS1 __attribute__((address_space(1)))
#define AS3 __attribute__((address_space(3)))

#if defined(__has_builtin)
#  if __has_builtin(__builtin_amdgcn_global_load_async_to_lds_b128)
#    define HAVE_ASYNC_B128 1
#  endif
#  if __has_builtin(__builtin_amdgcn_s_wait_asynccnt)
#    define HAVE_WAIT_ASYNC 1
#  endif
#endif

namespace {
constexpr int kB = 4, kP = 16, kIH = 512, kIW = 512;
constexpr int kE = 512, kH = 8, kDepth = 4, kL = 64, kK = 1024;
constexpr int kSH = kIH / kP, kSW = kIW / kP;   // 32, 32
constexpr int kN  = kSH * kSW;                  // 1024 tokens
constexpr int kPD = 3 * kP * kP;                // 768
constexpr int kHD = kE / kH;                    // 64
constexpr int kRD = 32;                         // rotary dims
constexpr int kBN = kB * kN;                    // 4096
constexpr long long kRecSz = (long long)kB * 3 * kIH * kIW;
constexpr float kPi = 3.14159265358979323846f;
} // namespace

// ---------------------------------------------------------------------------
// CDNA5 async global->LDS copy (ASYNCcnt-tracked), 16 bytes per lane.
// Builtin prototype (probe-confirmed by diagnostics): (v4i AS1*, v4i AS3*, Ii, Ii)
// ---------------------------------------------------------------------------
__device__ __forceinline__ void async_ld_b128(_Float16* l, const _Float16* g) {
#ifdef HAVE_ASYNC_B128
  __builtin_amdgcn_global_load_async_to_lds_b128(
      (AS1 v4i*)(AS1 const void*)g, (AS3 v4i*)(AS3 void*)l, 0, 0);
#else
  const unsigned lo = (unsigned)(unsigned long long)(AS3 void*)l;
  asm volatile("global_load_async_to_lds_b128 %0, %1, off" :: "v"(lo), "v"(g) : "memory");
#endif
}

template <int N>
__device__ __forceinline__ void wait_async() {
#ifdef HAVE_WAIT_ASYNC
  __builtin_amdgcn_s_wait_asynccnt((unsigned short)N);
#else
  if constexpr (N == 0) asm volatile("s_wait_asynccnt 0x0" ::: "memory");
  else                  asm volatile("s_wait_asynccnt 0x3" ::: "memory");
#endif
}

// ---------------------------------------------------------------------------
// WMMA fragment load from an LDS tile laid out [rows][LDK halfs], row-major.
// CDNA5 16-bit A-matrix 16x32 layout (ISA 7.12.2):
//   lanes 0-15 : M=lane,    VGPR0..3 = K 0..7 (2/VGPR), VGPR4..7 = K 16..23
//   lanes 16-31: M=lane-16, VGPR0..3 = K 8..15,          VGPR4..7 = K 24..31
// ---------------------------------------------------------------------------
__device__ __forceinline__ v16h ldfrag(const _Float16* __restrict__ base, int ld, int lane) {
  const int g = (lane >> 4) & 1;
  const int m = lane & 15;
  const _Float16* r = base + m * ld;
  v16h f;
#pragma unroll
  for (int j = 0; j < 4; ++j) {
    f[2 * j]         = r[8 * g + 2 * j];
    f[2 * j + 1]     = r[8 * g + 2 * j + 1];
    f[8 + 2 * j]     = r[16 + 8 * g + 2 * j];
    f[8 + 2 * j + 1] = r[16 + 8 * g + 2 * j + 1];
  }
  return f;
}

// ---------------------------------------------------------------------------
// f16 WMMA GEMM:  out = act(scale * A @ B^T + bias) (+ res)
//   A  : [batch][M, K] f16 row-major     B : [batch][N, K] f16 row-major
//   C  : optional f32 [batch][M,N]       C16: optional f16 [batch][M,N]
// 256 threads (8 wave32), 128x64 tile; wave = 32x32 via 2x2 WMMA; K step 32.
// Double-buffered LDS staged with GLOBAL_LOAD_ASYNC_TO_LDS_B128 (3 async
// wave-instructions per stage), drained with s_wait_asynccnt.
// Requires M%128==0, N%64==0, K%32==0 (true for all GEMMs in this model).
// ---------------------------------------------------------------------------
template <int ACT>
__global__ __launch_bounds__(256) void k_gemm16(
    const _Float16* __restrict__ A, const _Float16* __restrict__ Bm,
    const float* __restrict__ bias, const float* __restrict__ res,
    float* __restrict__ C, _Float16* __restrict__ C16,
    int Nn, int Kk, long long sA, long long sB, long long sC, float scale) {
  constexpr int TM = 128, TN = 64, TK = 32, LDK = TK + 8; // pad: 16B-aligned rows
  __shared__ __align__(16) _Float16 As[2][TM * LDK];
  __shared__ __align__(16) _Float16 Bs[2][TN * LDK];

  const int tid  = threadIdx.x;
  const int lane = tid & 31;
  const int wave = tid >> 5;
  const int wm   = (wave >> 1) * 32;   // 0..96
  const int wn   = (wave & 1) * 32;    // 0..32
  const int gn0  = blockIdx.x * TN;
  const int gm0  = blockIdx.y * TM;
  const _Float16* Ab = A + (long long)blockIdx.z * sA;
  const _Float16* Bb = Bm + (long long)blockIdx.z * sB;

  auto stage = [&](int buf, int k0) {
    // A tile: 128 rows x 4 chunks of 8 halfs -> 2 async instructions/thread
#pragma unroll
    for (int p = 0; p < 2; ++p) {
      const int ia = tid + p * 256;
      const int r = ia >> 2, c = ia & 3;
      async_ld_b128(&As[buf][r * LDK + c * 8],
                    Ab + (long long)(gm0 + r) * Kk + (k0 + c * 8));
    }
    // B tile: 64 rows x 4 chunks -> 1 async instruction/thread
    {
      const int r = tid >> 2, c = tid & 3;
      async_ld_b128(&Bs[buf][r * LDK + c * 8],
                    Bb + (long long)(gn0 + r) * Kk + (k0 + c * 8));
    }
  };

  v8f acc[2][2];
#pragma unroll
  for (int i = 0; i < 2; ++i)
#pragma unroll
    for (int j = 0; j < 2; ++j)
#pragma unroll
      for (int r = 0; r < 8; ++r) acc[i][j][r] = 0.0f;

  stage(0, 0);
  int buf = 0;
  for (int k0 = 0; k0 < Kk; k0 += TK) {
    const bool more = (k0 + TK) < Kk;
    if (more) {
      stage(buf ^ 1, k0 + TK);
      wait_async<3>();   // newest 3 may be in flight; current stage complete
    } else {
      wait_async<0>();
    }
    __syncthreads();
    const v16h a0 = ldfrag(&As[buf][(wm + 0) * LDK], LDK, lane);
    const v16h a1 = ldfrag(&As[buf][(wm + 16) * LDK], LDK, lane);
    const v16h b0 = ldfrag(&Bs[buf][(wn + 0) * LDK], LDK, lane);
    const v16h b1 = ldfrag(&Bs[buf][(wn + 16) * LDK], LDK, lane);
    acc[0][0] = __builtin_amdgcn_wmma_f32_16x16x32_f16(false, a0, false, b0, (short)0, acc[0][0], false, false);
    acc[0][1] = __builtin_amdgcn_wmma_f32_16x16x32_f16(false, a0, false, b1, (short)0, acc[0][1], false, false);
    acc[1][0] = __builtin_amdgcn_wmma_f32_16x16x32_f16(false, a1, false, b0, (short)0, acc[1][0], false, false);
    acc[1][1] = __builtin_amdgcn_wmma_f32_16x16x32_f16(false, a1, false, b1, (short)0, acc[1][1], false, false);
    __syncthreads();
    buf ^= 1;
  }

  // Epilogue per 16x16 f32 C/D layout: VGPR r -> M = r + 8*(lane>=16), N = lane&15
  const int g = lane >> 4, nn = lane & 15;
  float* Cb = C ? C + (long long)blockIdx.z * sC : nullptr;
  _Float16* C16b = C16 ? C16 + (long long)blockIdx.z * sC : nullptr;
#pragma unroll
  for (int i = 0; i < 2; ++i)
#pragma unroll
    for (int j = 0; j < 2; ++j) {
      const int col = gn0 + wn + j * 16 + nn;
      const float bv = bias ? bias[col] : 0.0f;
#pragma unroll
      for (int r = 0; r < 8; ++r) {
        const int row = gm0 + wm + i * 16 + g * 8 + r;
        float v = acc[i][j][r] * scale + bv;
        if (ACT == 1) v = 0.5f * v * (1.0f + erff(v * 0.70710678118654752f));
        const long long idx = (long long)row * Nn + col;
        if (res) v += res[idx];
        if (Cb) Cb[idx] = v;
        if (C16b) C16b[idx] = (_Float16)v;
      }
    }
}

// ---------------------------------------------------------------------------
// Elementwise / reduction kernels
// ---------------------------------------------------------------------------
__global__ void k_cvt16(const float* __restrict__ in, _Float16* __restrict__ out, long long n) {
  const long long i = (long long)blockIdx.x * blockDim.x + threadIdx.x;
  if (i < n) out[i] = (_Float16)in[i];
}

__global__ void k_patchify(const float* __restrict__ x, _Float16* __restrict__ pt) {
  const long long i = (long long)blockIdx.x * blockDim.x + threadIdx.x;
  if (i >= (long long)kBN * kPD) return;
  const int pd = (int)(i % kPD);
  const int n  = (int)((i / kPD) % kN);
  const int b  = (int)(i / ((long long)kPD * kN));
  const int c = pd / (kP * kP), ph = (pd / kP) % kP, pw = pd % kP;
  const int sh = n / kSW, sw = n % kSW;
  pt[i] = (_Float16)x[(((long long)(b * 3 + c) * kIH) + sh * kP + ph) * kIW + sw * kP + pw];
}

// LayerNorm over E=512, block=256 (2 elems/thread); optional +pos_emb.
// Writes f32 (residual stream) and/or f16 (next-GEMM operand).
__global__ void k_layernorm(const float* __restrict__ x, const float* __restrict__ w,
                            const float* __restrict__ b, const float* __restrict__ pos,
                            float* __restrict__ o32, _Float16* __restrict__ o16) {
  __shared__ float red[16];
  const int row = blockIdx.x;
  const int t = threadIdx.x;
  const float* xr = x + (long long)row * kE;
  const float v0 = xr[t], v1 = xr[t + 256];
  float s = v0 + v1, s2 = v0 * v0 + v1 * v1;
#pragma unroll
  for (int off = 16; off > 0; off >>= 1) {
    s  += __shfl_xor(s, off, 32);
    s2 += __shfl_xor(s2, off, 32);
  }
  if ((t & 31) == 0) { red[t >> 5] = s; red[8 + (t >> 5)] = s2; }
  __syncthreads();
  if (t < 32) {
    float a = (t < 8) ? red[t] : 0.0f;
    float c = (t < 8) ? red[8 + t] : 0.0f;
#pragma unroll
    for (int off = 4; off > 0; off >>= 1) {
      a += __shfl_xor(a, off, 32);
      c += __shfl_xor(c, off, 32);
    }
    if (t == 0) { red[0] = a; red[1] = c; }
  }
  __syncthreads();
  const float mean = red[0] / (float)kE;
  const float var  = red[1] / (float)kE - mean * mean;
  const float rstd = rsqrtf(var + 1e-5f);
  const int n = row % kN;
  const float p0 = pos ? pos[(long long)n * kE + t] : 0.0f;
  const float p1 = pos ? pos[(long long)n * kE + t + 256] : 0.0f;
  const float r0 = (v0 - mean) * rstd * w[t] + b[t] + p0;
  const float r1 = (v1 - mean) * rstd * w[t + 256] + b[t + 256] + p1;
  if (o32) { o32[(long long)row * kE + t] = r0; o32[(long long)row * kE + t + 256] = r1; }
  if (o16) { o16[(long long)row * kE + t] = (_Float16)r0; o16[(long long)row * kE + t + 256] = (_Float16)r1; }
}

// qkv (B,N,3E) f32 -> q,k (B,H,N,HD) f16 with axial RoPE; v transposed (B,H,HD,N) f16.
__global__ void k_qkv_rope(const float* __restrict__ qkv, _Float16* __restrict__ q,
                           _Float16* __restrict__ k, _Float16* __restrict__ vt) {
  const long long i = (long long)blockIdx.x * blockDim.x + threadIdx.x;
  if (i >= (long long)kB * kH * kN * kHD) return;
  const int d = (int)(i % kHD);
  const int n = (int)((i / kHD) % kN);
  const int h = (int)((i / ((long long)kHD * kN)) % kH);
  const int b = (int)(i / ((long long)kHD * kN * kH));
  const long long base = ((long long)(b * kN + n)) * (3 * kE) + h * kHD + d;
  const float qe = qkv[base], ke = qkv[base + kE], ve = qkv[base + 2 * kE];
  float qo = qe, ko = ke;
  if (d < kRD) {
    const int sh = n / kSW, sw = n % kSW;
    const float band = (1.0f + (float)((d & 15) >> 1) * (511.0f / 7.0f)) * kPi;
    const float lin = (d < 16) ? (-1.0f + 2.0f * (float)sh / (float)(kSH - 1))
                               : (-1.0f + 2.0f * (float)sw / (float)(kSW - 1));
    const float f = lin * band;
    const float cs = cosf(f), sn = sinf(f);
    const int partner = d ^ 1;
    const float sgn = (d & 1) ? 1.0f : -1.0f;
    const float qp = qkv[base - d + partner];
    const float kp = qkv[base - d + partner + kE];
    qo = qe * cs + sgn * qp * sn;
    ko = ke * cs + sgn * kp * sn;
  }
  q[i] = (_Float16)qo;
  k[i] = (_Float16)ko;
  vt[(((long long)(b * kH + h)) * kHD + d) * kN + n] = (_Float16)ve;
}

// Row softmax over 1024 cols; writes f32 in place + f16 copy for the AV GEMM.
__global__ void k_softmax(float* __restrict__ s, _Float16* __restrict__ s16) {
  __shared__ float red[8];
  const long long row = blockIdx.x;
  float* sr = s + row * (long long)kN;
  _Float16* hr = s16 + row * (long long)kN;
  const int t = threadIdx.x;
  float v[4];
  float mx = -3.402823466e38f;
#pragma unroll
  for (int j = 0; j < 4; ++j) { v[j] = sr[t + j * 256]; mx = fmaxf(mx, v[j]); }
#pragma unroll
  for (int off = 16; off > 0; off >>= 1) mx = fmaxf(mx, __shfl_xor(mx, off, 32));
  if ((t & 31) == 0) red[t >> 5] = mx;
  __syncthreads();
  if (t < 32) {
    float a = (t < 8) ? red[t] : -3.402823466e38f;
#pragma unroll
    for (int off = 4; off > 0; off >>= 1) a = fmaxf(a, __shfl_xor(a, off, 32));
    if (t == 0) red[0] = a;
  }
  __syncthreads();
  mx = red[0];
  float sum = 0.0f;
#pragma unroll
  for (int j = 0; j < 4; ++j) { v[j] = expf(v[j] - mx); sum += v[j]; }
#pragma unroll
  for (int off = 16; off > 0; off >>= 1) sum += __shfl_xor(sum, off, 32);
  __syncthreads();
  if ((t & 31) == 0) red[t >> 5] = sum;
  __syncthreads();
  if (t < 32) {
    float a = (t < 8) ? red[t] : 0.0f;
#pragma unroll
    for (int off = 4; off > 0; off >>= 1) a += __shfl_xor(a, off, 32);
    if (t == 0) red[0] = a;
  }
  __syncthreads();
  const float inv = 1.0f / red[0];
#pragma unroll
  for (int j = 0; j < 4; ++j) {
    const float o = v[j] * inv;
    sr[t + j * 256] = o;
    hr[t + j * 256] = (_Float16)o;
  }
}

// (B,H,N,HD) f16 -> (B,N,E) f16
__global__ void k_merge(const _Float16* __restrict__ in, _Float16* __restrict__ out) {
  const long long i = (long long)blockIdx.x * blockDim.x + threadIdx.x;
  if (i >= (long long)kBN * kE) return;
  const int d = (int)(i % kHD);
  const int n = (int)((i / kHD) % kN);
  const int h = (int)((i / ((long long)kHD * kN)) % kH);
  const int b = (int)(i / ((long long)kHD * kN * kH));
  out[((long long)(b * kN + n)) * kE + h * kHD + d] = in[i];
}

__global__ void k_add_pos(float* __restrict__ x, const float* __restrict__ pos) {
  const long long i = (long long)blockIdx.x * blockDim.x + threadIdx.x;
  if (i >= (long long)kBN * kE) return;
  x[i] += pos[i % ((long long)kN * kE)];
}

// L2-normalize rows of width 64; optional f32 and f16 outputs.
__global__ void k_norm_rows(const float* __restrict__ z, float* __restrict__ o32,
                            _Float16* __restrict__ o16, int rows) {
  const int r = blockIdx.x * blockDim.x + threadIdx.x;
  if (r >= rows) return;
  const float* zr = z + (long long)r * kL;
  float s = 1e-12f;
  for (int i = 0; i < kL; ++i) s += zr[i] * zr[i];
  const float inv = rsqrtf(s);
  for (int i = 0; i < kL; ++i) {
    const float v = zr[i] * inv;
    if (o32) o32[(long long)r * kL + i] = v;
    if (o16) o16[(long long)r * kL + i] = (_Float16)v;
  }
}

__global__ void k_zero(float* p) {
  if (blockIdx.x == 0 && threadIdx.x == 0) *p = 0.0f;
}

// Per-row argmax over K=1024 sims, gather codebook row (f16 for decoder GEMM),
// accumulate commit loss and emit index.
__global__ void k_vq(const float* __restrict__ sim, const float* __restrict__ cbn,
                     const float* __restrict__ z, _Float16* __restrict__ qv,
                     float* __restrict__ loss, float* __restrict__ idx_out) {
  __shared__ float bv[256];
  __shared__ int bi[256];
  __shared__ float lred[2];
  const int r = blockIdx.x, t = threadIdx.x;
  const float* sr = sim + (long long)r * kK;
  float best = -3.402823466e38f;
  int bidx = kK;
  for (int c = t; c < kK; c += 256) {
    const float vv = sr[c];
    if (vv > best || (vv == best && c < bidx)) { best = vv; bidx = c; }
  }
  bv[t] = best; bi[t] = bidx;
  __syncthreads();
  for (int off = 128; off > 0; off >>= 1) {
    if (t < off) {
      if (bv[t + off] > bv[t] || (bv[t + off] == bv[t] && bi[t + off] < bi[t])) {
        bv[t] = bv[t + off];
        bi[t] = bi[t + off];
      }
    }
    __syncthreads();
  }
  const int idx = bi[0];
  float pl = 0.0f;
  if (t < kL) {
    const float qc = cbn[(long long)idx * kL + t];
    qv[(long long)r * kL + t] = (_Float16)qc;
    const float df = qc - z[(long long)r * kL + t];
    pl = df * df;
  }
#pragma unroll
  for (int off = 16; off > 0; off >>= 1) pl += __shfl_xor(pl, off, 32);
  if (t < kL && (t & 31) == 0) lred[t >> 5] = pl;
  __syncthreads();
  if (t == 0) {
    atomicAdd(loss, (lred[0] + lred[1]) * (0.25f / ((float)kBN * (float)kL)));
    idx_out[r] = (float)idx;
  }
}

__global__ void k_unpatch_sig(const float* __restrict__ patches, float* __restrict__ out) {
  const long long i = (long long)blockIdx.x * blockDim.x + threadIdx.x;
  if (i >= kRecSz) return;
  const int iw = (int)(i % kIW);
  const int ih = (int)((i / kIW) % kIH);
  const int c  = (int)((i / ((long long)kIW * kIH)) % 3);
  const int b  = (int)(i / (3LL * kIW * kIH));
  const int sh = ih / kP, ph = ih % kP, sw = iw / kP, pw = iw % kP;
  const long long src = ((long long)(b * kN + sh * kSW + sw)) * kPD + (c * kP * kP + ph * kP + pw);
  out[i] = 1.0f / (1.0f + expf(-patches[src]));
}

// ---------------------------------------------------------------------------
// Host-side helpers
// ---------------------------------------------------------------------------
static inline void gemm16(hipStream_t st, const _Float16* A, const _Float16* Bm,
                          const float* bias, const float* res,
                          float* C, _Float16* C16,
                          int M, int Nn, int Kk, int batch,
                          long long sA, long long sB, long long sC,
                          float scale, int act) {
  dim3 g(Nn / 64, M / 128, batch), blk(256);
  if (act)
    k_gemm16<1><<<g, blk, 0, st>>>(A, Bm, bias, res, C, C16, Nn, Kk, sA, sB, sC, scale);
  else
    k_gemm16<0><<<g, blk, 0, st>>>(A, Bm, bias, res, C, C16, Nn, Kk, sA, sB, sC, scale);
}

static inline void cvt(hipStream_t st, const float* in, _Float16* out, long long n) {
  k_cvt16<<<(unsigned)((n + 255) / 256), 256, 0, st>>>(in, out, n);
}

extern "C" void kernel_launch(void* const* d_in, const int* in_sizes, int n_in,
                              void* d_out_v, int out_size, void* d_ws, size_t ws_size,
                              hipStream_t stream) {
  (void)in_sizes; (void)n_in; (void)out_size; (void)ws_size;
  const float* x        = (const float*)d_in[0];
  const float* patch_w  = (const float*)d_in[1];
  const float* patch_b  = (const float*)d_in[2];
  const float* pe_ln_w  = (const float*)d_in[3];
  const float* pe_ln_b  = (const float*)d_in[4];
  const float* pos_emb  = (const float*)d_in[5];
  const float* enc_p[12]; const float* dec_p[12];
  for (int i = 0; i < 12; ++i) enc_p[i] = (const float*)d_in[6 + i];
  for (int i = 0; i < 12; ++i) dec_p[i] = (const float*)d_in[18 + i];
  const float* enc_norm_w = (const float*)d_in[30];
  const float* enc_norm_b = (const float*)d_in[31];
  const float* quant_w    = (const float*)d_in[32];
  const float* quant_b    = (const float*)d_in[33];
  const float* codebook   = (const float*)d_in[34];
  const float* post_w     = (const float*)d_in[35];
  const float* post_b     = (const float*)d_in[36];
  const float* dec_norm_w = (const float*)d_in[37];
  const float* dec_norm_b = (const float*)d_in[38];
  const float* pred_w     = (const float*)d_in[39];
  const float* pred_b     = (const float*)d_in[40];
  float* out = (float*)d_out_v;

  char* wp = (char*)d_ws;
  auto alloc32 = [&](long long n) -> float* {
    float* r = (float*)wp; wp += ((n * 4 + 255) / 256) * 256; return r;
  };
  auto alloc16 = [&](long long n) -> _Float16* {
    _Float16* r = (_Float16*)wp; wp += ((n * 2 + 255) / 256) * 256; return r;
  };

  // f32 activations
  float* tok  = alloc32((long long)kBN * kE);
  float* lnb  = alloc32((long long)kBN * kE);
  float* qkv  = alloc32((long long)kBN * 3 * kE);
  float* sc   = alloc32((long long)kB * kH * kN * kN);
  float* zz   = alloc32((long long)kBN * kL);
  float* cbn  = alloc32((long long)kK * kL);
  float* sim  = alloc32((long long)kBN * kK);
  float* prd  = alloc32((long long)kBN * kPD);
  // f16 activations
  _Float16* pt16  = alloc16((long long)kBN * kPD);
  _Float16* ln16  = alloc16((long long)kBN * kE);
  _Float16* q16   = alloc16((long long)kBN * kE);
  _Float16* k16   = alloc16((long long)kBN * kE);
  _Float16* vt16  = alloc16((long long)kBN * kE);
  _Float16* sc16  = alloc16((long long)kB * kH * kN * kN);
  _Float16* ao16  = alloc16((long long)kBN * kE);
  _Float16* mh16  = alloc16((long long)kBN * kE);
  _Float16* ff16  = alloc16((long long)kBN * 4 * kE);
  _Float16* zn16  = alloc16((long long)kBN * kL);
  _Float16* cbn16 = alloc16((long long)kK * kL);
  _Float16* qv16  = alloc16((long long)kBN * kL);
  // f16 weights (converted once per call)
  _Float16* patch_w16 = alloc16((long long)kE * kPD);
  _Float16* quant_w16 = alloc16((long long)kL * kE);
  _Float16* post_w16  = alloc16((long long)kE * kL);
  _Float16* pred_w16  = alloc16((long long)kPD * kE);
  _Float16* qkv_w16[2], *proj_w16[2], *fc1_w16[2], *fc2_w16[2];
  for (int p = 0; p < 2; ++p) {
    qkv_w16[p]  = alloc16((long long)kDepth * 3 * kE * kE);
    proj_w16[p] = alloc16((long long)kDepth * kE * kE);
    fc1_w16[p]  = alloc16((long long)kDepth * 4 * kE * kE);
    fc2_w16[p]  = alloc16((long long)kDepth * kE * 4 * kE);
  }

  // ---- Weight conversion (f32 -> f16, once per forward) ----
  cvt(stream, patch_w, patch_w16, (long long)kE * kPD);
  cvt(stream, quant_w, quant_w16, (long long)kL * kE);
  cvt(stream, post_w,  post_w16,  (long long)kE * kL);
  cvt(stream, pred_w,  pred_w16,  (long long)kPD * kE);
  const float* const* pp[2] = {enc_p, dec_p};
  for (int p = 0; p < 2; ++p) {
    cvt(stream, pp[p][2],  qkv_w16[p],  (long long)kDepth * 3 * kE * kE);
    cvt(stream, pp[p][4],  proj_w16[p], (long long)kDepth * kE * kE);
    cvt(stream, pp[p][8],  fc1_w16[p],  (long long)kDepth * 4 * kE * kE);
    cvt(stream, pp[p][10], fc2_w16[p],  (long long)kDepth * kE * 4 * kE);
  }

  auto run_block = [&](int p, int i) {
    const float* ln1w = pp[p][0] + (long long)i * kE;
    const float* ln1b = pp[p][1] + (long long)i * kE;
    const float* qkvb = pp[p][3] + (long long)i * 3 * kE;
    const float* pb   = pp[p][5] + (long long)i * kE;
    const float* ln2w = pp[p][6] + (long long)i * kE;
    const float* ln2b = pp[p][7] + (long long)i * kE;
    const float* f1b  = pp[p][9] + (long long)i * 4 * kE;
    const float* f2b  = pp[p][11] + (long long)i * kE;
    const _Float16* qkvw = qkv_w16[p] + (long long)i * 3 * kE * kE;
    const _Float16* pw   = proj_w16[p] + (long long)i * kE * kE;
    const _Float16* f1w  = fc1_w16[p] + (long long)i * 4 * kE * kE;
    const _Float16* f2w  = fc2_w16[p] + (long long)i * kE * 4 * kE;

    k_layernorm<<<kBN, 256, 0, stream>>>(tok, ln1w, ln1b, nullptr, nullptr, ln16);
    gemm16(stream, ln16, qkvw, qkvb, nullptr, qkv, nullptr, kBN, 3 * kE, kE, 1, 0, 0, 0, 1.0f, 0);
    const long long tq = (long long)kB * kH * kN * kHD;
    k_qkv_rope<<<(unsigned)((tq + 255) / 256), 256, 0, stream>>>(qkv, q16, k16, vt16);
    gemm16(stream, q16, k16, nullptr, nullptr, sc, nullptr, kN, kN, kHD, kB * kH,
           (long long)kN * kHD, (long long)kN * kHD, (long long)kN * kN, 0.125f, 0);
    k_softmax<<<kB * kH * kN, 256, 0, stream>>>(sc, sc16);
    gemm16(stream, sc16, vt16, nullptr, nullptr, nullptr, ao16, kN, kHD, kN, kB * kH,
           (long long)kN * kN, (long long)kN * kHD, (long long)kN * kHD, 1.0f, 0);
    k_merge<<<(kBN * kE + 255) / 256, 256, 0, stream>>>(ao16, mh16);
    gemm16(stream, mh16, pw, pb, tok, tok, nullptr, kBN, kE, kE, 1, 0, 0, 0, 1.0f, 0);
    k_layernorm<<<kBN, 256, 0, stream>>>(tok, ln2w, ln2b, nullptr, nullptr, ln16);
    gemm16(stream, ln16, f1w, f1b, nullptr, nullptr, ff16, kBN, 4 * kE, kE, 1, 0, 0, 0, 1.0f, 1);
    gemm16(stream, ff16, f2w, f2b, tok, tok, nullptr, kBN, kE, 4 * kE, 1, 0, 0, 0, 1.0f, 0);
  };

  // ---- Patch embedding ----
  k_patchify<<<(kBN * kPD + 255) / 256, 256, 0, stream>>>(x, pt16);
  gemm16(stream, pt16, patch_w16, patch_b, nullptr, lnb, nullptr, kBN, kE, kPD, 1, 0, 0, 0, 1.0f, 0);
  k_layernorm<<<kBN, 256, 0, stream>>>(lnb, pe_ln_w, pe_ln_b, pos_emb, tok, nullptr);

  // ---- Encoder ----
  for (int i = 0; i < kDepth; ++i) run_block(0, i);
  k_layernorm<<<kBN, 256, 0, stream>>>(tok, enc_norm_w, enc_norm_b, nullptr, nullptr, ln16);

  // ---- Vector quantization ----
  gemm16(stream, ln16, quant_w16, quant_b, nullptr, zz, nullptr, kBN, kL, kE, 1, 0, 0, 0, 1.0f, 0);
  k_norm_rows<<<(kBN + 255) / 256, 256, 0, stream>>>(zz, nullptr, zn16, kBN);
  k_norm_rows<<<(kK + 255) / 256, 256, 0, stream>>>(codebook, cbn, cbn16, kK);
  gemm16(stream, zn16, cbn16, nullptr, nullptr, sim, nullptr, kBN, kK, kL, 1, 0, 0, 0, 1.0f, 0);
  k_zero<<<1, 1, 0, stream>>>(out + kRecSz);
  k_vq<<<kBN, 256, 0, stream>>>(sim, cbn, zz, qv16, out + kRecSz, out + kRecSz + 1);

  // ---- Decoder ----
  gemm16(stream, qv16, post_w16, post_b, nullptr, tok, nullptr, kBN, kE, kL, 1, 0, 0, 0, 1.0f, 0);
  k_add_pos<<<(kBN * kE + 255) / 256, 256, 0, stream>>>(tok, pos_emb);
  for (int i = 0; i < kDepth; ++i) run_block(1, i);
  k_layernorm<<<kBN, 256, 0, stream>>>(tok, dec_norm_w, dec_norm_b, nullptr, nullptr, ln16);
  gemm16(stream, ln16, pred_w16, pred_b, nullptr, prd, nullptr, kBN, kPD, kE, 1, 0, 0, 0, 1.0f, 0);
  k_unpatch_sig<<<(unsigned)((kRecSz + 255) / 256), 256, 0, stream>>>(prd, out);
}
